// CE2FlowOperator_12549894439359
// MI455X (gfx1250) — compile-verified, matrix operand order checked by reading, not run on hardware
//
#include <hip/hip_runtime.h>
#include <math.h>

typedef __attribute__((ext_vector_type(8)))  float  f32x8;
typedef __attribute__((ext_vector_type(16))) __bf16 bf16x16;
typedef __attribute__((ext_vector_type(8)))  __bf16 bf16x8;

#define HDIM 128
#define STEPS 10
#define ROWS_PER_WG 256
#define NTHREADS 512

#define A_STRIDE 136   // bf16 elems per row: 128 + 8 pad -> 272B row stride,
                       // bank(m) = 68m mod 64 = 4m : conflict-free A loads

// LDS byte offsets (dynamic shared memory; 320KB/WGP on CDNA5)
#define OFF_WE   0          // We packed:  4 ktiles x 16 ntiles x 1024B = 65536
#define OFF_WG   65536      // Wg packed:  4 x 8  x 1024B = 32768
#define OFF_WM1  98304      // Wm1 packed: 4 x 4  x 1024B = 16384
#define OFF_WD   114688     // Wd packed:  4 x 8  x 1024B = 32768
#define OFF_BIAS 147456     // 641 floats (be 256 | bg 128 | bm1 64 | bd 128 | wm2 64 | bm2 1)
#define OFF_A    150032     // 16 waves * (16*A_STRIDE*2 = 4352B) bf16 A slabs
#define A_SLAB_BYTES (16 * A_STRIDE * 2)
#define SMEM_BYTES (OFF_A + 16 * A_SLAB_BYTES)   // 219664 B

#define BI_BE  0
#define BI_BG  256
#define BI_BM1 384
#define BI_BD  448
#define BI_WM2 576
#define BI_BM2 640

// ---------- helpers ----------

// Pack a KxN fp32 weight matrix into bf16 WMMA B-operand fragments in LDS.
// Fragment (kt,nt) = 1024B, stored as two 512B half-blocks with 16B lane
// stride (2-way minimum bank conflicts for ds_load_b128):
//   lo block: elem (lane*8 + j),        j=0..7
//   hi block: elem (256 + lane*8 + j-8), j=8..15
// Logical element for (lane, j):
//   n = nt*16 + (lane&15); k = kt*32 + ((lane>=16)?8:0) + (j<8 ? j : j+8)
__device__ inline void pack_weights(const float* __restrict__ W, int K, int N,
                                    __bf16* __restrict__ dst, int tid, int nthreads) {
  const int ntiles = N >> 4;
  const int total  = (K >> 5) * ntiles * 512;
  for (int idx = tid; idx < total; idx += nthreads) {
    int frag = idx >> 9;
    int e    = idx & 511;
    int sub  = e >> 8;          // 0 = lo half-block, 1 = hi half-block
    int r    = e & 255;
    int lane = r >> 3;
    int j    = sub * 8 + (r & 7);
    int kt   = frag / ntiles;
    int nt   = frag - kt * ntiles;
    int n    = nt * 16 + (lane & 15);
    int k    = kt * 32 + ((lane >> 4) << 3) + ((j < 8) ? j : j + 8);
    dst[idx] = (__bf16)W[k * N + n];
  }
}

// baseL = packed + lane*8 (hoisted, loop-invariant). frag index is a
// compile-time constant after unrolling -> pure DS instruction offsets.
__device__ inline bf16x16 load_b_frag(const __bf16* __restrict__ baseL, int frag) {
  const __bf16* base = baseL + frag * 512;
  bf16x8 lo = *(const bf16x8*)(base);
  bf16x8 hi = *(const bf16x8*)(base + 256);
  return __builtin_shufflevector(lo, hi, 0,1,2,3,4,5,6,7,8,9,10,11,12,13,14,15);
}

// aBaseL = Aw + (lane&15)*A_STRIDE + ((lane>>4)&1)*8 (hoisted).
__device__ inline bf16x16 load_a_frag(const __bf16* __restrict__ aBaseL, int kt) {
  const __bf16* base = aBaseL + kt * 32;
  bf16x8 lo = *(const bf16x8*)(base);
  bf16x8 hi = *(const bf16x8*)(base + 16);
  return __builtin_shufflevector(lo, hi, 0,1,2,3,4,5,6,7,8,9,10,11,12,13,14,15);
}

// sBaseL = Aw + (lane>>4)*8*A_STRIDE + (lane&15) (hoisted).
__device__ inline void store_frag_bf16(__bf16* __restrict__ sBaseL, int j, f32x8 v) {
  #pragma unroll
  for (int r = 0; r < 8; ++r)
    sBaseL[r * A_STRIDE + j * 16] = (__bf16)v[r];
}

__device__ inline f32x8 gemm_tile(const bf16x16* __restrict__ a,
                                  const __bf16* __restrict__ baseL,
                                  int ntiles, int nt, float bias) {
  f32x8 acc = { bias, bias, bias, bias, bias, bias, bias, bias };
  #pragma unroll
  for (int kt = 0; kt < 4; ++kt) {
    bf16x16 b = load_b_frag(baseL, kt * ntiles + nt);
    acc = __builtin_amdgcn_wmma_f32_16x16x32_bf16(
        false, a[kt], false, b, (short)0, acc, false, false);
  }
  return acc;
}

__device__ inline void wait_ds() {
  asm volatile("s_wait_dscnt 0" ::: "memory");
}

// ---------- kernel ----------

// LDS (215KB) pins occupancy at 1 WG/WGP = 4 waves/SIMD anyway; ask for the
// maximum VGPR budget so nothing spills to scratch.
__global__ void __launch_bounds__(NTHREADS, 1)
ce2_flow_kernel(const float* __restrict__ x,
                const float* __restrict__ We,  const float* __restrict__ be,
                const float* __restrict__ Wg,  const float* __restrict__ bg,
                const float* __restrict__ Wm1, const float* __restrict__ bm1,
                const float* __restrict__ Wm2, const float* __restrict__ bm2,
                const float* __restrict__ Wd,  const float* __restrict__ bd,
                float* __restrict__ out) {
  extern __shared__ char smem[];
  float* bias = (float*)(smem + OFF_BIAS);

  const int tid    = threadIdx.x;
  const int lane   = tid & 31;
  const int waveId = tid >> 5;
  const int c      = lane & 15;
  const int hb     = lane >> 4;

  // ---- one-time: pack weights (bf16, B-operand layout) + biases into LDS ----
  pack_weights(We,  128, 256, (__bf16*)(smem + OFF_WE),  tid, NTHREADS);
  pack_weights(Wg,  128, 128, (__bf16*)(smem + OFF_WG),  tid, NTHREADS);
  pack_weights(Wm1, 128,  64, (__bf16*)(smem + OFF_WM1), tid, NTHREADS);
  pack_weights(Wd,  128, 128, (__bf16*)(smem + OFF_WD),  tid, NTHREADS); // first 128 rows
  for (int i = tid; i < 256; i += NTHREADS) bias[BI_BE  + i] = be[i];
  for (int i = tid; i < 128; i += NTHREADS) bias[BI_BG  + i] = bg[i];
  for (int i = tid; i <  64; i += NTHREADS) bias[BI_BM1 + i] = bm1[i];
  for (int i = tid; i < 128; i += NTHREADS) bias[BI_BD  + i] = bd[i];
  for (int i = tid; i <  64; i += NTHREADS) bias[BI_WM2 + i] = Wm2[i];
  if (tid == 0) bias[BI_BM2] = bm2[0];
  __syncthreads();

  // ---- hoisted per-lane LDS base pointers (all hot accesses become
  //      base + compile-time DS instruction offsets; nothing to spill) ----
  __bf16* Aw = (__bf16*)(smem + OFF_A + waveId * A_SLAB_BYTES);
  const __bf16* bWe  = (const __bf16*)(smem + OFF_WE)  + lane * 8;
  const __bf16* bWg  = (const __bf16*)(smem + OFF_WG)  + lane * 8;
  const __bf16* bWm1 = (const __bf16*)(smem + OFF_WM1) + lane * 8;
  const __bf16* bWd  = (const __bf16*)(smem + OFF_WD)  + lane * 8;
  const __bf16* aLd  = Aw + c * A_STRIDE + hb * 8;   // A-fragment load base
  __bf16*       aSt  = Aw + hb * 8 * A_STRIDE + c;   // A-fragment store base
  const float*  biasC = bias + c;                     // per-lane bias base

  const int rowBase = blockIdx.x * ROWS_PER_WG + waveId * 16;

  // ---- load this wave's 16x128 fp32 state tile into C-layout fragments ----
  f32x8 s[8];
  #pragma unroll
  for (int j = 0; j < 8; ++j) {
    #pragma unroll
    for (int r = 0; r < 8; ++r)
      s[j][r] = x[(size_t)(rowBase + r + hb * 8) * HDIM + j * 16 + c];
  }

  const float bm2v = bias[BI_BM2];

  #pragma unroll 1
  for (int step = 0; step < STEPS; ++step) {
    bf16x16 a[4];

    // phase 1: state -> bf16 A slab; encoder + gate + flow update
    #pragma unroll
    for (int j = 0; j < 8; ++j) store_frag_bf16(aSt, j, s[j]);
    wait_ds();
    #pragma unroll
    for (int kt = 0; kt < 4; ++kt) a[kt] = load_a_frag(aLd, kt);

    #pragma unroll
    for (int j = 0; j < 8; ++j) {
      f32x8 g  = gemm_tile(a, bWg, 8,  j,     biasC[BI_BG + j * 16]);
      f32x8 mg = gemm_tile(a, bWe, 16, j,     biasC[BI_BE + j * 16]);
      f32x8 dr = gemm_tile(a, bWe, 16, j + 8, biasC[BI_BE + 128 + j * 16]);
      #pragma unroll
      for (int r = 0; r < 8; ++r) {
        float gv = 1.0f / (1.0f + __expf(-g[r]));
        s[j][r] += (mg[r] * gv) * tanhf(dr[r] * gv) * 0.1f;
      }
    }

    // phase 2: new -> A slab; mirror detector (WMMA 128->64, VALU dot 64->1)
    #pragma unroll
    for (int j = 0; j < 8; ++j) store_frag_bf16(aSt, j, s[j]);
    wait_ds();
    #pragma unroll
    for (int kt = 0; kt < 4; ++kt) a[kt] = load_a_frag(aLd, kt);

    float hsum[8];
    #pragma unroll
    for (int r = 0; r < 8; ++r) hsum[r] = 0.0f;
    #pragma unroll
    for (int jt = 0; jt < 4; ++jt) {
      f32x8 h = gemm_tile(a, bWm1, 4, jt, biasC[BI_BM1 + jt * 16]);
      float w2 = biasC[BI_WM2 + jt * 16];
      #pragma unroll
      for (int r = 0; r < 8; ++r) hsum[r] += fmaxf(h[r], 0.0f) * w2;
    }
    bool mirr[8];
    #pragma unroll
    for (int r = 0; r < 8; ++r) {
      float t = hsum[r];
      t += __shfl_xor(t, 1, 32);
      t += __shfl_xor(t, 2, 32);
      t += __shfl_xor(t, 4, 32);
      t += __shfl_xor(t, 8, 32);
      mirr[r] = (t + bm2v) > 0.0f;          // sigmoid(z)>0.5  <=>  z>0
    }

    // phase 3: conditional flip along cols (pure lane permute, fp32)
    #pragma unroll
    for (int j = 0; j < 4; ++j) {
      int k = 7 - j;
      #pragma unroll
      for (int r = 0; r < 8; ++r) {
        float vj = s[j][r], vk = s[k][r];
        float fj = __shfl_xor(vk, 15, 32);   // partner: frag 7-j, lane^0xF
        float fk = __shfl_xor(vj, 15, 32);
        s[j][r] = mirr[r] ? fj : vj;
        s[k][r] = mirr[r] ? fk : vk;
      }
    }

    // phase 4: mirrored -> A slab; decoder GEMM writes next state
    #pragma unroll
    for (int j = 0; j < 8; ++j) store_frag_bf16(aSt, j, s[j]);
    wait_ds();
    #pragma unroll
    for (int kt = 0; kt < 4; ++kt) a[kt] = load_a_frag(aLd, kt);
    #pragma unroll
    for (int j = 0; j < 8; ++j)
      s[j] = gemm_tile(a, bWd, 8, j, biasC[BI_BD + j * 16]);
  }

  // ---- write fp32 result ----
  #pragma unroll
  for (int j = 0; j < 8; ++j) {
    #pragma unroll
    for (int r = 0; r < 8; ++r)
      out[(size_t)(rowBase + r + hb * 8) * HDIM + j * 16 + c] = s[j][r];
  }
}

extern "C" void kernel_launch(void* const* d_in, const int* in_sizes, int n_in,
                              void* d_out, int out_size, void* d_ws, size_t ws_size,
                              hipStream_t stream) {
  const float* x   = (const float*)d_in[0];
  const float* We  = (const float*)d_in[1];
  const float* be  = (const float*)d_in[2];
  const float* Wg  = (const float*)d_in[3];
  const float* bg  = (const float*)d_in[4];
  const float* Wm1 = (const float*)d_in[5];
  const float* bm1 = (const float*)d_in[6];
  const float* Wm2 = (const float*)d_in[7];
  const float* bm2 = (const float*)d_in[8];
  const float* Wd  = (const float*)d_in[9];
  const float* bd  = (const float*)d_in[10];
  float* out = (float*)d_out;

  const int rows = in_sizes[0] / HDIM;          // B*S = 131072
  const int grid = rows / ROWS_PER_WG;          // 512 workgroups

  ce2_flow_kernel<<<grid, NTHREADS, SMEM_BYTES, stream>>>(
      x, We, be, Wg, bg, Wm1, bm1, Wm2, bm2, Wd, bd, out);
}
